// TwoCompRecurrentSNN_9990093930924
// MI455X (gfx1250) — matrix operand order repaired and checked by reading, not compiled, not run on hardware
//
#include <hip/hip_runtime.h>

// ---------------------------------------------------------------------------
// Two-compartment recurrent SNN, T=100 sequential steps.
// Matmuls in bf16 WMMA (v_wmma_f32_16x16x32_bf16), f32 accumulate.
// 4 kernels per step; no atomics; weights pre-packed into WMMA fragments.
// ---------------------------------------------------------------------------

typedef __attribute__((ext_vector_type(16))) __bf16 v16bf;
typedef __attribute__((ext_vector_type(8)))  __bf16 v8bf;
typedef __attribute__((ext_vector_type(8)))  float  v8f;

#define T_STEPS 100
#define BATCH   256
#define N_IN    784
#define N_IN_P  800      // padded to 25 * 32
#define N_HID   1024
#define N_OUT   10
#define SLOTS   8        // circular delay slots (power of 2, > MAX_DELAY+1)

#define ALPHA_M   0.951229424500714f     // exp(-0.05)
#define ALPHA_B   0.9048374180359595f    // exp(-0.1)
#define ALPHA_A   0.951229424500714f     // exp(-0.05)
#define ALPHA_PRE 0.8187307530779818f    // exp(-0.2)
#define V_TH      0.2f
#define V_RESET   0.0f
#define ETA_SCALED (0.0005f / (256.0f * 256.0f))  // ETA_IN / (B*B)

// -------------------------------- WMMA helpers -----------------------------

__device__ __forceinline__ v8f wmma_bf16(v16bf a, v16bf b, v8f c) {
  return __builtin_amdgcn_wmma_f32_16x16x32_bf16(
      /*neg_a=*/false, a, /*neg_b=*/false, b,
      /*c_mod=*/(short)0, c, /*reuse_a=*/false, /*reuse_b=*/false);
}

// A fragment (16x32 bf16, row-major source). base points at row 0 of the
// 16-row tile. Lane layout: m = lane&15, half = lane>>4;
//   elems 0..15 = A[m][kBase + half*16 + 0..15]
__device__ __forceinline__ v16bf load_a_frag(const __bf16* base, int ld,
                                             int kBase, int lane) {
  const int m = lane & 15, half = lane >> 4;
  const __bf16* p = base + (size_t)m * ld + kBase + half * 16;
  v8bf lo = *(const v8bf*)(p);
  v8bf hi = *(const v8bf*)(p + 8);
  v16bf a;
#pragma unroll
  for (int i = 0; i < 8; ++i) { a[i] = lo[i]; a[i + 8] = hi[i]; }
  return a;
}

// B fragment: pre-packed, 512 bf16 per 32x16 tile, lane-major (lane*16 + e).
__device__ __forceinline__ v16bf load_b_frag(const __bf16* packed, int tile,
                                             int lane) {
  return *(const v16bf*)(packed + (size_t)tile * 512 + lane * 16);
}

// -------------------------------- setup kernels ----------------------------

__global__ void snn_zero_ws(unsigned int* p, size_t nwords) {
  size_t i = (size_t)blockIdx.x * blockDim.x + threadIdx.x;
  size_t stride = (size_t)gridDim.x * blockDim.x;
  for (; i < nwords; i += stride) p[i] = 0u;
}

// Copy W_in [784,1024] into padded fp32 master [800,1024] (pad rows zero).
__global__ void snn_copy_win(const float* __restrict__ src, float* dst) {
  size_t i = (size_t)blockIdx.x * blockDim.x + threadIdx.x;
  size_t stride = (size_t)gridDim.x * blockDim.x;
  for (; i < (size_t)N_IN_P * N_HID; i += stride) {
    int r = (int)(i / N_HID);
    dst[i] = (r < N_IN) ? src[(size_t)r * N_HID + (i % N_HID)] : 0.0f;
  }
}

// Pack an fp32 row-major [K, ld] matrix into bf16 WMMA B-fragment tiles.
// Tile (kt, nt) -> dst[(kt*nTiles+nt)*512 + lane*16 + e], where for lane l:
//   n = l&15, half = l>>4, element e = src[kt*32 + half*16 + e][nt*16 + n]
__global__ void snn_pack_b(const float* __restrict__ src, __bf16* dst,
                           int kTotal, int nTotal, int ld, int kTiles,
                           int nTiles) {
  const int lane = threadIdx.x & 31;
  const int wv = blockIdx.x * (blockDim.x >> 5) + (threadIdx.x >> 5);
  if (wv >= kTiles * nTiles) return;
  const int kt = wv / nTiles, nt = wv % nTiles;
  const int n = lane & 15, half = lane >> 4;
  const int col = nt * 16 + n;
  __bf16* out = dst + (size_t)wv * 512 + lane * 16;
#pragma unroll
  for (int e = 0; e < 16; ++e) {
    const int k = kt * 32 + half * 16 + e;
    float v = (k < kTotal && col < nTotal) ? src[(size_t)k * ld + col] : 0.0f;
    out[e] = (__bf16)v;
  }
}

// -------------------------------- per-step kernels -------------------------

// Fused: I_b = x_del@W_in + h_b@W_rec_b, I_a = h_a@W_rec_a, neuron update,
// spike generation + store, dendritic-error partial sums (no atomics).
// Grid: 64 blocks x 256 threads (8 waves); wave -> 16x32 hidden tile
// (1 M-tile x 2 N-tiles, A fragment reused across both N tiles).
__global__ void __launch_bounds__(256)
snn_gemm_update(const __bf16* __restrict__ xdel,   // [256][800]  x_{t-1}
                const __bf16* __restrict__ hb,     // [256][1024] s_{t-3}
                const __bf16* __restrict__ ha,     // [256][1024] s_{t-4}
                const __bf16* __restrict__ pWin,   // packed 25x64 tiles
                const __bf16* __restrict__ pWrb,   // packed 32x64 tiles
                const __bf16* __restrict__ pWra,   // packed 32x64 tiles
                float* __restrict__ u, float* __restrict__ ub,
                float* __restrict__ ua,
                __bf16* __restrict__ sOut,         // [256][1024] slot for s_t
                float* __restrict__ dend_part) {   // [32][1024] partials
  const int lane = threadIdx.x & 31;
  const int w = blockIdx.x * 8 + (threadIdx.x >> 5);  // 0..511
  const int mt = w >> 5;          // 0..15 (batch tiles)
  const int nt0 = (w & 31) * 2;   // 0..62 (first of two hidden tiles)
  const int rowBase = mt * 16;

  v8f acc_b0 = {0.f, 0.f, 0.f, 0.f, 0.f, 0.f, 0.f, 0.f};
  v8f acc_b1 = acc_b0, acc_a0 = acc_b0, acc_a1 = acc_b0;

  const __bf16* aIn = xdel + (size_t)rowBase * N_IN_P;
  const __bf16* aHb = hb + (size_t)rowBase * N_HID;
  const __bf16* aHa = ha + (size_t)rowBase * N_HID;

  // feed-forward: x_del @ W_in   (K = 800)
  for (int kt = 0; kt < 25; ++kt) {
    v16bf a = load_a_frag(aIn, N_IN_P, kt * 32, lane);
    v16bf b0 = load_b_frag(pWin, kt * 64 + nt0, lane);
    v16bf b1 = load_b_frag(pWin, kt * 64 + nt0 + 1, lane);
    if (kt + 1 < 25)
      __builtin_prefetch(pWin + (size_t)((kt + 1) * 64 + nt0) * 512, 0, 1);
    acc_b0 = wmma_bf16(a, b0, acc_b0);
    acc_b1 = wmma_bf16(a, b1, acc_b1);
  }
  // recurrent basal: h_b @ W_rec_b   (K = 1024)
  for (int kt = 0; kt < 32; ++kt) {
    v16bf a = load_a_frag(aHb, N_HID, kt * 32, lane);
    v16bf b0 = load_b_frag(pWrb, kt * 64 + nt0, lane);
    v16bf b1 = load_b_frag(pWrb, kt * 64 + nt0 + 1, lane);
    if (kt + 1 < 32)
      __builtin_prefetch(pWrb + (size_t)((kt + 1) * 64 + nt0) * 512, 0, 1);
    acc_b0 = wmma_bf16(a, b0, acc_b0);
    acc_b1 = wmma_bf16(a, b1, acc_b1);
  }
  // recurrent apical: h_a @ W_rec_a   (K = 1024)
  for (int kt = 0; kt < 32; ++kt) {
    v16bf a = load_a_frag(aHa, N_HID, kt * 32, lane);
    v16bf b0 = load_b_frag(pWra, kt * 64 + nt0, lane);
    v16bf b1 = load_b_frag(pWra, kt * 64 + nt0 + 1, lane);
    if (kt + 1 < 32)
      __builtin_prefetch(pWra + (size_t)((kt + 1) * 64 + nt0) * 512, 0, 1);
    acc_a0 = wmma_bf16(a, b0, acc_a0);
    acc_a1 = wmma_bf16(a, b1, acc_a1);
  }

  // epilogue: leaky integration, threshold, reset, spike store, dend partials
  const int half = lane >> 4;
  const int n = lane & 15;
#pragma unroll
  for (int tile = 0; tile < 2; ++tile) {
    const v8f accb = tile ? acc_b1 : acc_b0;
    const v8f acca = tile ? acc_a1 : acc_a0;
    const int col = (nt0 + tile) * 16 + n;
    float dlocal = 0.0f;
#pragma unroll
    for (int r = 0; r < 8; ++r) {
      const int row = rowBase + r + 8 * half;
      const size_t idx = (size_t)row * N_HID + col;
      float ubn = ALPHA_B * ub[idx] + (1.0f - ALPHA_B) * accb[r];
      float uan = ALPHA_A * ua[idx] + (1.0f - ALPHA_A) * acca[r];
      float un = ALPHA_M * u[idx] + (1.0f - ALPHA_M) * (ubn + uan);
      float s = (un >= V_TH) ? 1.0f : 0.0f;
      un = (s > 0.0f) ? V_RESET : un;
      ub[idx] = ubn;
      ua[idx] = uan;
      u[idx] = un;
      sOut[idx] = (__bf16)s;
      dlocal += (uan - ubn);
    }
    // unique (mt, half, col) writer -> no atomics, fully overwritten per step
    dend_part[(size_t)(mt * 2 + half) * N_HID + col] = dlocal;
  }
}

// Fused: presynaptic trace update + column sums, and injection of x_t into
// the input delay line (slot t; everything else this step reads slot t-1).
// One block per input column j; pre stored transposed [784][256].
__global__ void __launch_bounds__(256)
snn_trace_inject(const __bf16* __restrict__ xdel,  // slot t-1
                 const float* __restrict__ xt,     // x_spikes[t], [256][784]
                 __bf16* __restrict__ slot_xt,     // slot t, [256][800]
                 float* __restrict__ pre_t, float* __restrict__ pre_sum) {
  __shared__ float red[256];
  const int j = blockIdx.x;       // 0..783
  const int b = threadIdx.x;      // 0..255
  // inject x_t (padding columns 784..799 stay zero from setup forever)
  slot_xt[(size_t)b * N_IN_P + j] = (__bf16)xt[(size_t)b * N_IN + j];
  // trace update on x_{t-1}
  float xd = (float)xdel[(size_t)b * N_IN_P + j];
  float p = ALPHA_PRE * pre_t[(size_t)j * BATCH + b] + xd;
  pre_t[(size_t)j * BATCH + b] = p;
  red[b] = p;
  __syncthreads();
  for (int s = 128; s > 0; s >>= 1) {
    if (b < s) red[b] += red[b + s];
    __syncthreads();
  }
  if (b == 0) pre_sum[j] = red[0];
}

// out[t] = s_t @ W_out + b_out via WMMA (N padded 10 -> 16), plus the
// 32-way reduction of dend partials -> dend_sum (512 threads, 1024 cols).
// Grid: 2 blocks x 8 waves; wave = batch tile mt.
__global__ void __launch_bounds__(256)
snn_wout(const __bf16* __restrict__ s, const __bf16* __restrict__ pWout,
         const float* __restrict__ b_out, float* __restrict__ out,
         const float* __restrict__ dend_part, float* __restrict__ dend_sum) {
  const int lane = threadIdx.x & 31;
  const int mt = blockIdx.x * 8 + (threadIdx.x >> 5);  // 0..15
  const __bf16* aS = s + (size_t)mt * 16 * N_HID;
  v8f acc = {0.f, 0.f, 0.f, 0.f, 0.f, 0.f, 0.f, 0.f};
  for (int kt = 0; kt < 32; ++kt) {
    v16bf a = load_a_frag(aS, N_HID, kt * 32, lane);
    v16bf b = load_b_frag(pWout, kt, lane);
    acc = wmma_bf16(a, b, acc);
  }
  const int n = lane & 15, half = lane >> 4;
  if (n < N_OUT) {
    const float bias = b_out[n];
#pragma unroll
    for (int r = 0; r < 8; ++r) {
      const int row = mt * 16 + r + 8 * half;
      out[(size_t)row * N_OUT + n] = acc[r] + bias;
    }
  }
  // dendritic-error reduction: dend_sum[c] = sum_{p<32} dend_part[p][c]
  const int gtid = blockIdx.x * 256 + threadIdx.x;  // 0..511
  for (int c = gtid; c < N_HID; c += 512) {
    float sum = 0.0f;
#pragma unroll
    for (int p = 0; p < 32; ++p) sum += dend_part[(size_t)p * N_HID + c];
    dend_sum[c] = sum;
  }
}

// Fused plasticity + repack: W_in += ETA/(B*B) * outer(pre_sum, dend_sum),
// writing both the fp32 master and the bf16 WMMA fragment location.
__global__ void __launch_bounds__(256)
snn_win_update_pack(float* __restrict__ win, __bf16* __restrict__ pWin,
                    const float* __restrict__ pre_sum,
                    const float* __restrict__ dend_sum) {
  const size_t i = (size_t)blockIdx.x * blockDim.x + threadIdx.x;
  if (i >= (size_t)N_IN * N_HID) return;
  const int r = (int)(i / N_HID), c = (int)(i % N_HID);
  float v = win[(size_t)r * N_HID + c] +
            ETA_SCALED * pre_sum[r] * dend_sum[c];
  win[(size_t)r * N_HID + c] = v;
  // packed fragment index for row r, col c (matches snn_pack_b layout)
  const int kt = r >> 5, half = (r >> 4) & 1, e = r & 15;
  const int nt = c >> 4, n = c & 15, lane = half * 16 + n;
  pWin[((size_t)(kt * 64 + nt) * 512) + lane * 16 + e] = (__bf16)v;
}

// -------------------------------- host launch ------------------------------

extern "C" void kernel_launch(void* const* d_in, const int* in_sizes, int n_in,
                              void* d_out, int out_size, void* d_ws,
                              size_t ws_size, hipStream_t stream) {
  (void)in_sizes; (void)n_in; (void)out_size; (void)ws_size;
  const float* x_spikes = (const float*)d_in[0];  // [100,256,784]
  const float* W_in     = (const float*)d_in[1];  // [784,1024]
  const float* W_rec_b  = (const float*)d_in[2];  // [1024,1024]
  const float* W_rec_a  = (const float*)d_in[3];  // [1024,1024]
  const float* W_out    = (const float*)d_in[4];  // [1024,10]
  const float* b_out    = (const float*)d_in[5];  // [10]
  float* out = (float*)d_out;                     // [100,256,10]

  // --- workspace layout (256B aligned slabs) ---
  char* w = (char*)d_ws;
  size_t off = 0;
  auto take = [&](size_t bytes) {
    size_t r = off;
    off += (bytes + 255) & ~(size_t)255;
    return r;
  };
  __bf16* hist_in   = (__bf16*)(w + take((size_t)SLOTS * BATCH * N_IN_P * 2));
  __bf16* hist_hid  = (__bf16*)(w + take((size_t)SLOTS * BATCH * N_HID * 2));
  float*  u         = (float*) (w + take((size_t)BATCH * N_HID * 4));
  float*  ub        = (float*) (w + take((size_t)BATCH * N_HID * 4));
  float*  ua        = (float*) (w + take((size_t)BATCH * N_HID * 4));
  float*  pre_t     = (float*) (w + take((size_t)N_IN * BATCH * 4));
  float*  win_f32   = (float*) (w + take((size_t)N_IN_P * N_HID * 4));
  __bf16* pWin      = (__bf16*)(w + take((size_t)25 * 64 * 512 * 2));
  __bf16* pWrb      = (__bf16*)(w + take((size_t)32 * 64 * 512 * 2));
  __bf16* pWra      = (__bf16*)(w + take((size_t)32 * 64 * 512 * 2));
  __bf16* pWout     = (__bf16*)(w + take((size_t)32 * 1 * 512 * 2));
  float*  dend_part = (float*) (w + take((size_t)32 * N_HID * 4));
  float*  dend_sum  = (float*) (w + take((size_t)N_HID * 4));
  float*  pre_sum   = (float*) (w + take((size_t)N_IN * 4));
  const size_t total_bytes = off;

  // --- setup (run every call: deterministic) ---
  snn_zero_ws<<<2048, 256, 0, stream>>>((unsigned int*)d_ws, total_bytes / 4);
  snn_copy_win<<<1024, 256, 0, stream>>>(W_in, win_f32);
  snn_pack_b<<<(32 * 64 + 7) / 8, 256, 0, stream>>>(W_rec_b, pWrb, N_HID,
                                                    N_HID, N_HID, 32, 64);
  snn_pack_b<<<(32 * 64 + 7) / 8, 256, 0, stream>>>(W_rec_a, pWra, N_HID,
                                                    N_HID, N_HID, 32, 64);
  snn_pack_b<<<(32 + 7) / 8, 256, 0, stream>>>(W_out, pWout, N_HID, N_OUT,
                                               N_OUT, 32, 1);
  snn_pack_b<<<(25 * 64 + 7) / 8, 256, 0, stream>>>(win_f32, pWin, N_IN_P,
                                                    N_HID, N_HID, 25, 64);

  // --- time loop: sequential scan, 4 kernels per step chained on stream ---
  for (int t = 0; t < T_STEPS; ++t) {
    const int s_t  = t & (SLOTS - 1);
    const int s_x1 = (t + SLOTS - 1) & (SLOTS - 1);  // x_{t-1}
    const int s_h3 = (t + SLOTS - 3) & (SLOTS - 1);  // s_{t-3} (basal)
    const int s_h4 = (t + SLOTS - 4) & (SLOTS - 1);  // s_{t-4} (apical)

    __bf16* slot_xt = hist_in + (size_t)s_t * BATCH * N_IN_P;
    __bf16* slot_xd = hist_in + (size_t)s_x1 * BATCH * N_IN_P;
    __bf16* slot_hb = hist_hid + (size_t)s_h3 * BATCH * N_HID;
    __bf16* slot_ha = hist_hid + (size_t)s_h4 * BATCH * N_HID;
    __bf16* slot_st = hist_hid + (size_t)s_t * BATCH * N_HID;

    snn_gemm_update<<<64, 256, 0, stream>>>(slot_xd, slot_hb, slot_ha, pWin,
                                            pWrb, pWra, u, ub, ua, slot_st,
                                            dend_part);
    snn_trace_inject<<<N_IN, 256, 0, stream>>>(
        slot_xd, x_spikes + (size_t)t * BATCH * N_IN, slot_xt, pre_t, pre_sum);
    snn_wout<<<2, 256, 0, stream>>>(slot_st, pWout, b_out,
                                    out + (size_t)t * BATCH * N_OUT,
                                    dend_part, dend_sum);
    snn_win_update_pack<<<(N_IN * N_HID + 255) / 256, 256, 0, stream>>>(
        win_f32, pWin, pre_sum, dend_sum);
  }
}
